// ModelAttention_13829794693746
// MI455X (gfx1250) — matrix operand order repaired and checked by reading, not compiled, not run on hardware
//
#include <hip/hip_runtime.h>
#include <hip/hip_bf16.h>

#define Bb 16
#define Ss 2048
#define Hh 1024

typedef __attribute__((ext_vector_type(16))) __bf16 v16bf;
typedef __attribute__((ext_vector_type(8)))  float  v8f;

union FragBF {
    uint4 q[2];
    v16bf v;
};

// f32 -> bf16 (round-to-nearest-even-ish) as raw bits; avoids relying on __bf16 conversions
__device__ __forceinline__ unsigned short f2bf(float f) {
    unsigned int u = __builtin_bit_cast(unsigned int, f);
    u += 0x7FFFu + ((u >> 16) & 1u);
    return (unsigned short)(u >> 16);
}

// ---------------------------------------------------------------------------
// Kernel A: pack Wa = attn_W[:, H:2H] into bf16, k-contiguous per output row o.
// Wt stored as uints (2 bf16 per uint): Wt_u[o*512 + k/2]
// grid: 1024 x 256 ; each thread converts one float4 (4 elements)
__global__ void pack_wa_kernel(const float* __restrict__ attn_W,
                               unsigned int* __restrict__ Wt_u) {
    int idx = blockIdx.x * blockDim.x + threadIdx.x;   // 0 .. 262143
    int o  = idx >> 8;        // 256 float4 per row
    int c4 = idx & 255;
    float4 w = *(const float4*)(attn_W + (size_t)o * (2 * Hh) + Hh + c4 * 4);
    unsigned int lo = (unsigned int)f2bf(w.x) | ((unsigned int)f2bf(w.y) << 16);
    unsigned int hi = (unsigned int)f2bf(w.z) | ((unsigned int)f2bf(w.w) << 16);
    uint2 p; p.x = lo; p.y = hi;
    *(uint2*)(Wt_u + (size_t)o * 512 + c4 * 2) = p;
}

// ---------------------------------------------------------------------------
// Kernel B: bias[b*H + o] = attn_b[o] + dot(h[b,:], attn_W[o, 0:H])
// grid: 64 x 256, one thread per (b,o)
__global__ void bias_kernel(const float* __restrict__ h,
                            const float* __restrict__ attn_W,
                            const float* __restrict__ attn_b,
                            float* __restrict__ bias) {
    int idx = blockIdx.x * blockDim.x + threadIdx.x;   // = b*1024 + o
    int b = idx >> 10;
    int o = idx & 1023;
    const float4* hp = (const float4*)(h + (size_t)b * Hh);
    const float4* wp = (const float4*)(attn_W + (size_t)o * (2 * Hh));
    float acc = 0.f;
#pragma unroll 4
    for (int i = 0; i < Hh / 4; ++i) {
        float4 hv = hp[i];
        float4 wv = wp[i];
        acc += hv.x * wv.x + hv.y * wv.y + hv.z * wv.z + hv.w * wv.w;
    }
    bias[idx] = acc + attn_b[o];
}

// ---------------------------------------------------------------------------
// Kernel C: fused scores GEMM.
// One workgroup = 8 waves handles one (b, 16-row s-tile); waves split the o-axis.
// scores[b*S + s] = sum_o relu( a[b,s,:]·Wa[o,:] + bias[b,o] ) * v[o]
#define APAD_U 2                         // 2 uints = 4 halves pad per row
#define AROW_U (Hh / 2 + APAD_U + 2)     // 516 uints per LDS row (bank-spread)
__global__ void __launch_bounds__(256)
score_gemm_kernel(const float* __restrict__ a,
                  const unsigned int* __restrict__ Wt_u,
                  const float* __restrict__ bias,
                  const float* __restrict__ vW,
                  float* __restrict__ scores) {
    __shared__ unsigned int As_u[16 * AROW_U];   // 16 rows x 1024 bf16 (padded) ~33KB
    __shared__ float sp_lds[16];

    const int tid  = threadIdx.x;
    const int blk  = blockIdx.x;          // 0 .. B*S/16-1
    const int b    = blk >> 7;            // / (S/16)
    const int s0   = (blk & 127) << 4;    // tile base row

    if (tid < 16) sp_lds[tid] = 0.f;

    // ---- stage a[b, s0:s0+16, :] into LDS as bf16 (coalesced float4 reads) ----
    {
        const float* abase = a + ((size_t)b * Ss + s0) * Hh;
#pragma unroll
        for (int i = 0; i < 16; ++i) {
            int it  = tid + i * 256;       // 0 .. 4095
            int row = it >> 8;             // 256 float4 per row
            int c4  = it & 255;
            float4 av = *(const float4*)(abase + (size_t)row * Hh + c4 * 4);
            unsigned int lo = (unsigned int)f2bf(av.x) | ((unsigned int)f2bf(av.y) << 16);
            unsigned int hi = (unsigned int)f2bf(av.z) | ((unsigned int)f2bf(av.w) << 16);
            uint2 p; p.x = lo; p.y = hi;
            *(uint2*)(As_u + row * AROW_U + c4 * 2) = p;
        }
    }
    __syncthreads();

    const int lane = tid & 31;
    const int wave = tid >> 5;
    const int n    = lane & 15;     // column within 16x16 tile / row M for A
    const int hs   = lane >> 4;     // lane-group: A K-halves select / B K-group

    const unsigned int* arow = As_u + n * AROW_U;

    float sp[8];
#pragma unroll
    for (int r = 0; r < 8; ++r) sp[r] = 0.f;

    // each wave covers o in [wave*128, wave*128+128): 8 N-tiles
    for (int nt = 0; nt < 8; ++nt) {
        const int oc = wave * 128 + nt * 16 + n;                    // this lane's column o
        const unsigned int* wrow = Wt_u + (size_t)oc * 512 + hs * 4; // uints; halves oc*1024 + hs*8
        v8f acc = {};
        for (int k0 = 0; k0 < Hh; k0 += 32) {
            FragBF fa, fb;
            // A 16x32 bf16 frag (ISA 7.12.2): lane L holds row M=L%16,
            // elems 0..7 -> K = k0 + hs*8 + i ; elems 8..15 -> K = k0+16+hs*8+(i-8)
            int au = (k0 >> 1) + hs * 4;
            fa.q[0] = *(const uint4*)(arow + au);
            fa.q[1] = *(const uint4*)(arow + au + 8);
            // B 32x16 bf16 frag: lane L holds column N=L%16,
            // elems 0..7 -> K = k0 + hs*8 + i ; elems 8..15 -> K = k0+16+hs*8+(i-8)
            const unsigned int* wp = wrow + (k0 >> 1);
            fb.q[0] = *(const uint4*)(wp);
            fb.q[1] = *(const uint4*)(wp + 8);
            __builtin_prefetch(wp + 16, 0, 0);   // next K-chunk of weights
            acc = __builtin_amdgcn_wmma_f32_16x16x32_bf16(
                      false, fa.v, false, fb.v, (short)0, acc, false, false);
        }
        // fuse: relu(acc + bias[b,oc]) * v[oc]; acc[r] is (M = r + 8*hs, N = oc)
        float bv = bias[b * Hh + oc];
        float vv = vW[oc];
#pragma unroll
        for (int r = 0; r < 8; ++r) {
            float e = acc[r] + bv;
            e = e > 0.f ? e : 0.f;
            sp[r] += e * vv;
        }
    }

    // reduce partials across the 16 lanes of each half (sum over N within tiles)
#pragma unroll
    for (int r = 0; r < 8; ++r) {
        float v = sp[r];
        v += __shfl_xor(v, 1, 32);
        v += __shfl_xor(v, 2, 32);
        v += __shfl_xor(v, 4, 32);
        v += __shfl_xor(v, 8, 32);
        if (n == 0) atomicAdd(&sp_lds[r + 8 * hs], v);   // ds_add_f32
    }
    __syncthreads();
    if (tid < 16) scores[(size_t)b * Ss + s0 + tid] = sp_lds[tid];
}

// ---------------------------------------------------------------------------
// Kernel D: row softmax over s (one block per b), 256 threads x 8 elems
__global__ void softmax_kernel(const float* __restrict__ scores,
                               float* __restrict__ wout) {
    __shared__ float red[256];
    const int b = blockIdx.x;
    const int tid = threadIdx.x;
    float x[8];
#pragma unroll
    for (int i = 0; i < 8; ++i) x[i] = scores[(size_t)b * Ss + tid + i * 256];
    float m = x[0];
#pragma unroll
    for (int i = 1; i < 8; ++i) m = fmaxf(m, x[i]);
    red[tid] = m; __syncthreads();
    for (int s = 128; s > 0; s >>= 1) {
        if (tid < s) red[tid] = fmaxf(red[tid], red[tid + s]);
        __syncthreads();
    }
    const float M = red[0];
    __syncthreads();
    float sum = 0.f;
#pragma unroll
    for (int i = 0; i < 8; ++i) { x[i] = __expf(x[i] - M); sum += x[i]; }
    red[tid] = sum; __syncthreads();
    for (int s = 128; s > 0; s >>= 1) {
        if (tid < s) red[tid] += red[tid + s];
        __syncthreads();
    }
    const float inv = 1.f / red[0];
#pragma unroll
    for (int i = 0; i < 8; ++i) wout[(size_t)b * Ss + tid + i * 256] = x[i] * inv;
}

// ---------------------------------------------------------------------------
// Kernel E: context[b,h] = sum_s w[b,s] * a[b,s,h]   (grid 64 = 16 b x 4 h-chunks)
__global__ void context_kernel(const float* __restrict__ a,
                               const float* __restrict__ w,
                               float* __restrict__ ctx) {
    const int b = blockIdx.x >> 2;
    const int h = ((blockIdx.x & 3) << 8) + threadIdx.x;
    const float* ab = a + ((size_t)b * Ss) * Hh + h;
    const float* wr = w + (size_t)b * Ss;
    float acc = 0.f;
#pragma unroll 8
    for (int s = 0; s < Ss; ++s)
        acc += wr[s] * ab[(size_t)s * Hh];
    ctx[(size_t)b * Hh + h] = acc;
}

// ---------------------------------------------------------------------------
extern "C" void kernel_launch(void* const* d_in, const int* in_sizes, int n_in,
                              void* d_out, int out_size, void* d_ws, size_t ws_size,
                              hipStream_t stream) {
    const float* h      = (const float*)d_in[0];   // (1,B,H)
    /* c = d_in[1] unused by reference */
    const float* a      = (const float*)d_in[2];   // (B,S,H)
    const float* attn_W = (const float*)d_in[3];   // (H,2H)
    const float* attn_b = (const float*)d_in[4];   // (H,)
    const float* v_W    = (const float*)d_in[5];   // (1,H)

    float* ctx_out  = (float*)d_out;                       // B*H = 16384
    float* attn_out = (float*)d_out + (size_t)Bb * Hh;     // B*S = 32768

    char* ws = (char*)d_ws;
    unsigned int* Wt_u  = (unsigned int*)ws;                         // 2 MB bf16 weights
    float*        bias  = (float*)(ws + (2u << 20));                 // 64 KB
    float*        score = (float*)(ws + (2u << 20) + (64u << 10));   // 128 KB

    pack_wa_kernel<<<1024, 256, 0, stream>>>(attn_W, Wt_u);
    bias_kernel<<<64, 256, 0, stream>>>(h, attn_W, attn_b, bias);
    score_gemm_kernel<<<Bb * (Ss / 16), 256, 0, stream>>>(a, Wt_u, bias, v_W, score);
    softmax_kernel<<<Bb, 256, 0, stream>>>(score, attn_out);
    context_kernel<<<Bb * 4, 256, 0, stream>>>(a, attn_out, ctx_out);
}
